// PestDetectionSNN_86655260164621
// MI455X (gfx1250) — matrix operand order repaired and checked by reading, not compiled, not run on hardware
//
#include <hip/hip_runtime.h>
#include <hip/hip_bf16.h>
#include <stdint.h>

// ---------------------------------------------------------------------------
// Two-layer LIF SNN:
//   Phase 1: Z[B*T, H] = X[B*T, D] @ W1[D, H]   (bf16-split WMMA, f32 accum,
//            B operand staged into LDS per k-step via Tensor Data Mover)
//   Phase 2: sequential LIF scan over T per batch row (wave-level shuffle reduce)
// B=256, T=100, D=1024, H=128, C=5
// ---------------------------------------------------------------------------

#define B_  256
#define T_  100
#define D_  1024
#define H_  128
#define C_  5
#define M_  (B_ * T_)          // 25600 GEMM rows
#define KSTEPS (D_ / 32)       // 32 k-steps of K=32

#define DECAY1 0.90483741803596f   // exp(-1/10)
#define DECAY2 0.95122942450071f   // exp(-1/20)

typedef __bf16 bf16;
typedef __attribute__((ext_vector_type(16))) __bf16 v16bf;
typedef __attribute__((ext_vector_type(8)))  float  v8f;
typedef __attribute__((ext_vector_type(4)))  float  v4f;
typedef __attribute__((ext_vector_type(4)))  unsigned int u32x4;
typedef __attribute__((ext_vector_type(8)))  int    i32x8;
typedef __attribute__((ext_vector_type(4)))  int    i32x4;

// Per-k-step staged block: [hi/lo][8 n-tiles][32 lanes][16 bf16] = 16 KB
#define FRAG_ELEMS_PER_KS (2 * 8 * 32 * 16)     // 8192 bf16 = 16384 bytes
#define STAGE_BYTES       (FRAG_ELEMS_PER_KS * 2)

// Workspace layout (bytes)
#define Z_BYTES  ((size_t)M_ * H_ * 4)          // 13,107,200
#define WF_OFF   Z_BYTES                        // 512 KB pre-swizzled W (hi+lo)

// ---------------------------------------------------------------------------
// Kernel 0: pre-swizzle W1 (f32, row-major [D][H]) into per-lane WMMA B
// fragments, bf16 hi + bf16 lo, grouped so each k-step's block is a single
// contiguous 16 KB region:  Wf[ks][hl][nt][lane][16].
//
// B-fragment k order used here (contiguous, same as A; 32x16 K-major tile):
//   lane l: column n = l&15, k range klane = (l>>4)*16 .. +15,
//   element j of the 16-elem fragment = B[k0 + klane + j][n].
// ---------------------------------------------------------------------------
__global__ __launch_bounds__(32)
void snn_prep_w(const float* __restrict__ W1, bf16* __restrict__ wf) {
  const int lane = threadIdx.x;
  const int kb   = blockIdx.x >> 3;   // k-step (32 rows each)
  const int nt   = blockIdx.x & 7;    // n-tile (16 cols each)
  const int col  = nt * 16 + (lane & 15);
  const int kbase = kb * 32 + (lane >> 4) * 16;
  const size_t out_hi = ((size_t)((kb * 2 + 0) * 8 + nt) * 32 + lane) * 16;
  const size_t out_lo = ((size_t)((kb * 2 + 1) * 8 + nt) * 32 + lane) * 16;
#pragma unroll
  for (int j = 0; j < 16; ++j) {
    float w = W1[(size_t)(kbase + j) * H_ + col];
    bf16 hi = (bf16)w;
    wf[out_hi + j] = hi;
    wf[out_lo + j] = (bf16)(w - (float)hi);
  }
}

// ---------------------------------------------------------------------------
// TDM helper: one 1-D tensor DMA of `bytes` from global -> LDS.
// D# bitfields per CDNA5 ISA ch.8: group0 {count=1, lds_addr, global_addr,
// type=2}; group1 {data_size=1B, tensor_dim0=tile_dim0=bytes, stride=bytes}.
// This toolchain's builtin is the 6-arg form:
//   (u32x4 g0, i32x8 g1, i32x4 g2, i32x4 g3, i32x8 g4, i32 cpol)
// ---------------------------------------------------------------------------
#if __has_builtin(__builtin_amdgcn_tensor_load_to_lds)
#define HAVE_TDM 1
__device__ __forceinline__ void tdm_copy(const void* gsrc, uint32_t lds_addr,
                                         uint32_t bytes) {
  const uint64_t ga = (uint64_t)(uintptr_t)gsrc;
  u32x4 g0;
  g0[0] = 1u;                                            // count=1 (valid), user
  g0[1] = lds_addr;                                      // lds_addr[31:0]
  g0[2] = (uint32_t)ga;                                  // global_addr[31:0]
  g0[3] = (uint32_t)((ga >> 32) & 0x1FFFFFFu) | (2u << 30); // ga[56:32] | type=2
  i32x8 g1;
  g1[0] = 0;                                   // wg_mask=0, data_size=0 (1B)
  g1[1] = (int)((bytes & 0xFFFFu) << 16);      // tensor_dim0[15:0]
  g1[2] = (int)((bytes >> 16) | (1u << 16));   // tensor_dim0[31:16], tensor_dim1=1
  g1[3] = (int)((bytes & 0xFFFFu) << 16);      // tile_dim0 = bytes
  g1[4] = 0;                                   // tile_dim1=0, tile_dim2=0
  g1[5] = (int)bytes;                          // tensor_dim0_stride[31:0]
  g1[6] = 0;
  g1[7] = 0;
  const i32x4 zero4 = {0, 0, 0, 0};
  const i32x8 zero8 = {0, 0, 0, 0, 0, 0, 0, 0};
  __builtin_amdgcn_tensor_load_to_lds(g0, g1, zero4, zero4, zero8, 0);
}
#endif

__device__ __forceinline__ void wait_tensorcnt0() {
#if __has_builtin(__builtin_amdgcn_s_wait_tensorcnt)
  __builtin_amdgcn_s_wait_tensorcnt(0);
#else
  asm volatile("s_wait_tensorcnt 0x0" ::: "memory");
#endif
}

// ---------------------------------------------------------------------------
// Kernel 1: GEMM via v_wmma_f32_16x16x32_bf16, split-bf16 (hi*hi + lo*hi +
// hi*lo) for ~f32 accuracy. Each wave computes a 16(M) x 128(N) tile;
// 4 waves/block share the B operand staged in double-buffered LDS by the TDM.
// ---------------------------------------------------------------------------
__global__ __launch_bounds__(128)
void snn_gemm(const float* __restrict__ X, const bf16* __restrict__ wf,
              float* __restrict__ Z) {
  __shared__ __align__(16) bf16 smem[2 * FRAG_ELEMS_PER_KS];  // 32 KB, 2 buffers

  const int lane   = threadIdx.x & 31;
  const int wave   = threadIdx.x >> 5;
  const int m_tile = blockIdx.x * 4 + wave;
  const int m_base = m_tile * 16;

  // A-fragment addressing (contiguous k order, same as B fragments):
  //   lane l: row = l&15, klane = (l>>4)*16; elems 0..15 -> k = klane+0..15.
  const int  arow  = m_base + (lane & 15);
  const int  klane = (lane >> 4) * 16;
  const float* xrow = X + (size_t)arow * D_ + klane;

  v8f acc[8] = {};

  // ---- stage k-step 0 into buffer 0 ------------------------------------
#if defined(HAVE_TDM)
  if (wave == 0) {
    tdm_copy(wf, (uint32_t)(uintptr_t)&smem[0], STAGE_BYTES);
    wait_tensorcnt0();
  }
#else
  {
    const v16bf* src = (const v16bf*)wf;
    v16bf* dst = (v16bf*)&smem[0];
#pragma unroll
    for (int i = 0; i < 4; ++i)
      dst[threadIdx.x + i * 128] = src[threadIdx.x + i * 128];
  }
#endif
  __syncthreads();

  for (int ks = 0; ks < KSTEPS; ++ks) {
    const int buf = ks & 1;
    // ---- kick off DMA for next k-step into the other buffer ------------
#if defined(HAVE_TDM)
    if (wave == 0 && ks + 1 < KSTEPS) {
      tdm_copy(wf + (size_t)(ks + 1) * FRAG_ELEMS_PER_KS,
               (uint32_t)(uintptr_t)&smem[(buf ^ 1) * FRAG_ELEMS_PER_KS],
               STAGE_BYTES);
    }
#else
    if (ks + 1 < KSTEPS) {
      const v16bf* src = (const v16bf*)(wf + (size_t)(ks + 1) * FRAG_ELEMS_PER_KS);
      v16bf* dst = (v16bf*)&smem[(buf ^ 1) * FRAG_ELEMS_PER_KS];
#pragma unroll
      for (int i = 0; i < 4; ++i)
        dst[threadIdx.x + i * 128] = src[threadIdx.x + i * 128];
    }
#endif

    // ---- A: load 16 f32, split into bf16 hi/lo fragments ---------------
    const int k0 = ks * 32;
    __builtin_prefetch(xrow + k0 + 64, 0, 3);   // global_prefetch of A stream
    v4f x0 = *(const v4f*)(xrow + k0);
    v4f x1 = *(const v4f*)(xrow + k0 + 4);
    v4f x2 = *(const v4f*)(xrow + k0 + 8);
    v4f x3 = *(const v4f*)(xrow + k0 + 12);
    float xv[16];
#pragma unroll
    for (int i = 0; i < 4; ++i) {
      xv[i] = x0[i]; xv[4 + i] = x1[i]; xv[8 + i] = x2[i]; xv[12 + i] = x3[i];
    }
    v16bf ahi, alo;
#pragma unroll
    for (int i = 0; i < 16; ++i) {
      float f = xv[i];
      bf16 h = (bf16)f;
      ahi[i] = h;
      alo[i] = (bf16)(f - (float)h);
    }

    // ---- B from LDS: [buf][hl][nt][lane][16] ---------------------------
    const bf16* sb = &smem[buf * FRAG_ELEMS_PER_KS];
#pragma unroll
    for (int nt = 0; nt < 8; ++nt) {
      v16bf bhi = *(const v16bf*)(sb + (size_t)(0 * 8 + nt) * 512 + lane * 16);
      v16bf blo = *(const v16bf*)(sb + (size_t)(1 * 8 + nt) * 512 + lane * 16);
      acc[nt] = __builtin_amdgcn_wmma_f32_16x16x32_bf16(
          false, ahi, false, bhi, (short)0, acc[nt], false, false);
      acc[nt] = __builtin_amdgcn_wmma_f32_16x16x32_bf16(
          false, alo, false, bhi, (short)0, acc[nt], false, false);
      acc[nt] = __builtin_amdgcn_wmma_f32_16x16x32_bf16(
          false, ahi, false, blo, (short)0, acc[nt], false, false);
    }

#if defined(HAVE_TDM)
    if (wave == 0) wait_tensorcnt0();
#endif
    __syncthreads();
  }

  // C/D layout: VGPR r, lanes 0-15 -> row r, col = lane; lanes 16-31 -> row r+8.
  const int col  = lane & 15;
  const int rsel = (lane >> 4) * 8;
#pragma unroll
  for (int nt = 0; nt < 8; ++nt) {
#pragma unroll
    for (int r = 0; r < 8; ++r) {
      Z[(size_t)(m_base + rsel + r) * H_ + nt * 16 + col] = acc[nt][r];
    }
  }
}

// ---------------------------------------------------------------------------
// Kernel 2: sequential LIF scan. One wave32 per batch row; each lane owns
// 4 hidden units; s1@W2 reduced with shuffle-xor; v2/acc replicated per lane.
// ---------------------------------------------------------------------------
__global__ __launch_bounds__(32)
void snn_scan(const float* __restrict__ Z, const float* __restrict__ b1,
              const float* __restrict__ W2, const float* __restrict__ b2,
              float* __restrict__ out) {
  const int b    = blockIdx.x;
  const int lane = threadIdx.x;
  const int h0   = lane * 4;

  v4f b1v = *(const v4f*)(b1 + h0);
  float w2r[4][C_];
#pragma unroll
  for (int i = 0; i < 4; ++i)
#pragma unroll
    for (int c = 0; c < C_; ++c) w2r[i][c] = W2[(size_t)(h0 + i) * C_ + c];
  float b2v[C_];
#pragma unroll
  for (int c = 0; c < C_; ++c) b2v[c] = b2[c];

  float v1[4] = {0.f, 0.f, 0.f, 0.f};
  float v2[C_] = {};
  float accv[C_] = {};

  const float* zrow = Z + (size_t)b * T_ * H_ + h0;

  for (int t = 0; t < T_; ++t) {
    v4f z = *(const v4f*)(zrow + (size_t)t * H_);
    float contrib[C_] = {};
#pragma unroll
    for (int i = 0; i < 4; ++i) {
      float v = v1[i] * DECAY1 + z[i] + b1v[i];
      float s = (v >= 1.0f) ? 1.0f : 0.0f;
      v1[i] = v * (1.0f - s);
#pragma unroll
      for (int c = 0; c < C_; ++c) contrib[c] += s * w2r[i][c];
    }
    // wave32 xor-reduction: every lane ends with the full sum over H
#pragma unroll
    for (int off = 16; off >= 1; off >>= 1) {
#pragma unroll
      for (int c = 0; c < C_; ++c)
        contrib[c] += __shfl_xor(contrib[c], off, 32);
    }
#pragma unroll
    for (int c = 0; c < C_; ++c) {
      float v = v2[c] * DECAY2 + contrib[c] + b2v[c];
      float s = (v >= 1.0f) ? 1.0f : 0.0f;
      v2[c] = v * (1.0f - s);
      accv[c] += s;
    }
  }

  if (lane == 0) {
#pragma unroll
    for (int c = 0; c < C_; ++c) out[(size_t)b * C_ + c] = accv[c] * (1.0f / T_);
  }
}

// ---------------------------------------------------------------------------
extern "C" void kernel_launch(void* const* d_in, const int* in_sizes, int n_in,
                              void* d_out, int out_size, void* d_ws, size_t ws_size,
                              hipStream_t stream) {
  const float* X  = (const float*)d_in[0];  // [B, T, D]
  const float* W1 = (const float*)d_in[1];  // [D, H]
  const float* b1 = (const float*)d_in[2];  // [H]
  const float* W2 = (const float*)d_in[3];  // [H, C]
  const float* b2 = (const float*)d_in[4];  // [C]
  float* out = (float*)d_out;               // [B, C]

  char* ws = (char*)d_ws;
  float* Z  = (float*)ws;             // [M, H] f32 (fully overwritten)
  bf16* wf  = (bf16*)(ws + WF_OFF);   // pre-swizzled W fragments (hi+lo), 512 KB

  snn_prep_w<<<KSTEPS * (H_ / 16), 32, 0, stream>>>(W1, wf);
  snn_gemm<<<M_ / 16 / 4, 128, 0, stream>>>(X, wf, Z);
  snn_scan<<<B_, 32, 0, stream>>>(Z, b1, W2, b2, out);
}
